// STAttentionBlock_63136019251503
// MI455X (gfx1250) — matrix operand order repaired and checked by reading, not compiled, not run on hardware
//
#include <hip/hip_runtime.h>
#include <hip/hip_bf16.h>

// ST-Attention block for MI455X (gfx1250), wave32 + v_wmma_f32_16x16x32_bf16.
// Each wave computes a 32x32 macro-tile (2x2 WMMA tiles, 4 accumulators) so both
// operand fragments are amortized over two WMMAs each.
// Edge padding uses clamped addresses + value selects (no EXEC branching).

typedef __bf16 v16bf __attribute__((ext_vector_type(16)));
typedef float  v8f   __attribute__((ext_vector_type(8)));

// A-matrix (16x32 bf16) per-lane K index for vector element e.
// Lane = (m = lane&15, half = lane>>4). VGPR j<4: K = half*8 + 2j + lohi,
// VGPR j>=4: K = 16 + half*8 + 2(j-4) + lohi.  (ISA 7.12.2)
__device__ __forceinline__ int a_kpair(int j, int half) {
  return (j < 4) ? (half * 8 + j * 2) : (16 + half * 8 + (j - 4) * 2);
}
__device__ __forceinline__ int a_kmap(int e, int half) {
  return a_kpair(e >> 1, half) + (e & 1);
}

__device__ __forceinline__ v8f wmma_bf16(const v16bf& a, const v16bf& b, const v8f& c) {
  return __builtin_amdgcn_wmma_f32_16x16x32_bf16(false, a, false, b, (short)0, c, false, false);
}

// Branch-free tanh: 1 - 2/(e^{2x}+1); saturates correctly for |x| large.
__device__ __forceinline__ float tanh_fast(float x) {
  float e = __expf(2.f * x);
  return 1.f - 2.f / (e + 1.f);
}

// ---------------------------------------------------------------------------
// PE tables: pe_s[c][v] (64x25), pe_t[c][t] (64x32); computed once.
// ---------------------------------------------------------------------------
__device__ __forceinline__ float pe_val(int c, int pos) {
  float f = __expf(-(float)(c & ~1) * 0.14391157f);  // ln(1e4)/64
  float ang = (float)pos * f;
  return (c & 1) ? __cosf(ang) : __sinf(ang);
}
__global__ void make_pe(float* __restrict__ pe_s, float* __restrict__ pe_t) {
  int i = blockIdx.x * 256 + threadIdx.x;
  if (i < 1600) pe_s[i] = pe_val(i / 25, i % 25);
  if (i < 2048) pe_t[i] = pe_val(i / 32, i % 32);
}

// ---------------------------------------------------------------------------
// Generic batched conv1x1: out[n][o][p] = sum_c w[o][c]*in[n][c][p]  (p in [0,800))
// PE: 0 none, 1 spatial (pos=v), 2 temporal (pos=t).  BN: lrelu(res + bn(acc+bias)).
// 32x32 macro-tile per wave: 2 o-tiles x 2 p-tiles, 4 WMMAs per K-step.
// ---------------------------------------------------------------------------
template <int PE, int BN>
__global__ __launch_bounds__(32) void conv1x1_wmma(
    const float* __restrict__ in, const float* __restrict__ w,
    const float* __restrict__ bias, const float* __restrict__ res,
    const float* __restrict__ bn_g, const float* __restrict__ bn_b,
    const float* __restrict__ bn_m, const float* __restrict__ bn_v,
    const float* __restrict__ pe_tab,
    float* __restrict__ out, int K, int O) {
  const int lane = threadIdx.x & 31;
  const int half = lane >> 4, lm = lane & 15;
  const int pt = blockIdx.x * 2, ob = blockIdx.y * 32, n = blockIdx.z;
  const int p0 = pt * 16 + lm, p1 = p0 + 16;
  const float* inN = in + (size_t)n * (size_t)K * 800;
  const float* wrow0 = w + (size_t)(ob + lm) * K;
  const float* wrow1 = w + (size_t)(ob + 16 + lm) * K;
  const int PW = (PE == 1) ? 25 : 32;
  int pos0 = 0, pos1 = 0;
  if (PE) {
    pos0 = (PE == 1) ? (p0 % 25) : (p0 / 25);
    pos1 = (PE == 1) ? (p1 % 25) : (p1 / 25);
  }
  v8f acc00 = {}, acc01 = {}, acc10 = {}, acc11 = {};
  for (int k0 = 0; k0 < K; k0 += 32) {
    v16bf a0, a1, b0, b1;
#pragma unroll
    for (int j = 0; j < 8; ++j) {  // contiguous K-pairs -> b64 loads
      int kp = k0 + a_kpair(j, half);
      float2 w20 = *(const float2*)(wrow0 + kp);
      float2 w21 = *(const float2*)(wrow1 + kp);
      a0[2 * j] = (__bf16)w20.x; a0[2 * j + 1] = (__bf16)w20.y;
      a1[2 * j] = (__bf16)w21.x; a1[2 * j + 1] = (__bf16)w21.y;
    }
#pragma unroll
    for (int e = 0; e < 16; ++e) {
      int k = k0 + a_kmap(e, half);
      const float* row = inN + (size_t)k * 800;
      float v0 = row[p0], v1 = row[p1];
      if (PE) {
        const float* per = pe_tab + k * PW;
        v0 += per[pos0];
        v1 += per[pos1];
      }
      b0[e] = (__bf16)v0;
      b1[e] = (__bf16)v1;
    }
    acc00 = wmma_bf16(a0, b0, acc00);
    acc01 = wmma_bf16(a0, b1, acc01);
    acc10 = wmma_bf16(a1, b0, acc10);
    acc11 = wmma_bf16(a1, b1, acc11);
  }
  float* outN = out + (size_t)n * (size_t)O * 800;
  const float* resN = BN ? (res + (size_t)n * (size_t)O * 800) : nullptr;
#pragma unroll
  for (int r = 0; r < 8; ++r) {
    int o0 = ob + half * 8 + r, o1 = o0 + 16;
    float v00 = acc00[r] + bias[o0], v01 = acc01[r] + bias[o0];
    float v10 = acc10[r] + bias[o1], v11 = acc11[r] + bias[o1];
    if (BN) {
      float inv0 = bn_g[o0] * rsqrtf(bn_v[o0] + 1e-5f);
      float add0 = bn_b[o0] - bn_m[o0] * inv0;
      float inv1 = bn_g[o1] * rsqrtf(bn_v[o1] + 1e-5f);
      float add1 = bn_b[o1] - bn_m[o1] * inv1;
      v00 = v00 * inv0 + add0 + resN[(size_t)o0 * 800 + p0];
      v01 = v01 * inv0 + add0 + resN[(size_t)o0 * 800 + p1];
      v10 = v10 * inv1 + add1 + resN[(size_t)o1 * 800 + p0];
      v11 = v11 * inv1 + add1 + resN[(size_t)o1 * 800 + p1];
      v00 = v00 > 0.f ? v00 : 0.1f * v00;
      v01 = v01 > 0.f ? v01 : 0.1f * v01;
      v10 = v10 > 0.f ? v10 : 0.1f * v10;
      v11 = v11 > 0.f ? v11 : 0.1f * v11;
    }
    outN[(size_t)o0 * 800 + p0] = v00;
    outN[(size_t)o0 * 800 + p1] = v01;
    outN[(size_t)o1 * 800 + p0] = v10;
    outN[(size_t)o1 * 800 + p1] = v11;
  }
}

// ---------------------------------------------------------------------------
// Spatial attention: att[n][s][u][v] = tanh(sum_{c,t} q*k / 512)*alpha[s] + att0[s][u][v]
// K = (c,t) = 512; one wave does the whole padded 32x32 output (4 WMMAs/step).
// ---------------------------------------------------------------------------
__global__ __launch_bounds__(32) void att_spatial(
    const float* __restrict__ qk, const float* __restrict__ alphas,
    const float* __restrict__ att0, float* __restrict__ att) {
  const int lane = threadIdx.x & 31, half = lane >> 4, lm = lane & 15;
  const int n = blockIdx.x / 3, s = blockIdx.x % 3;
  const float* qb = qk + ((size_t)n * 96 + s * 16) * 800;
  const float* kb = qk + ((size_t)n * 96 + 48 + s * 16) * 800;
  const int u0 = lm, u1 = 16 + lm;           // u0 < 25 always
  const int v0 = lm, v1 = 16 + lm;
  const int u1c = u1 < 25 ? u1 : 24, v1c = v1 < 25 ? v1 : 24;
  const bool u1ok = u1 < 25, v1ok = v1 < 25;
  v8f acc00 = {}, acc01 = {}, acc10 = {}, acc11 = {};
  for (int k0 = 0; k0 < 512; k0 += 32) {
    v16bf a0, a1, b0, b1;
#pragma unroll
    for (int e = 0; e < 16; ++e) {
      int Ki = k0 + a_kmap(e, half);
      int c = Ki >> 5, t = Ki & 31;
      const float* qr = qb + (size_t)c * 800 + t * 25;
      a0[e] = (__bf16)qr[u0];
      float q1 = qr[u1c];
      a1[e] = (__bf16)(u1ok ? q1 : 0.f);
    }
#pragma unroll
    for (int e = 0; e < 16; ++e) {
      int Ki = k0 + a_kmap(e, half);
      int c = Ki >> 5, t = Ki & 31;
      const float* kr = kb + (size_t)c * 800 + t * 25;
      b0[e] = (__bf16)kr[v0];
      float k1 = kr[v1c];
      b1[e] = (__bf16)(v1ok ? k1 : 0.f);
    }
    acc00 = wmma_bf16(a0, b0, acc00);
    acc01 = wmma_bf16(a0, b1, acc01);
    acc10 = wmma_bf16(a1, b0, acc10);
    acc11 = wmma_bf16(a1, b1, acc11);
  }
  float alpha = alphas[s];
  float* ab = att + ((size_t)n * 3 + s) * 625;
#pragma unroll
  for (int r = 0; r < 8; ++r) {
    int uu = half * 8 + r, uu1 = 16 + uu;
    ab[uu * 25 + v0] =
        tanh_fast(acc00[r] * (1.f / 512.f)) * alpha + att0[(s * 25 + uu) * 25 + v0];
    if (v1ok)
      ab[uu * 25 + v1] =
          tanh_fast(acc01[r] * (1.f / 512.f)) * alpha + att0[(s * 25 + uu) * 25 + v1];
    if (uu1 < 25) {
      ab[uu1 * 25 + v0] =
          tanh_fast(acc10[r] * (1.f / 512.f)) * alpha + att0[(s * 25 + uu1) * 25 + v0];
      if (v1ok)
        ab[uu1 * 25 + v1] =
            tanh_fast(acc11[r] * (1.f / 512.f)) * alpha + att0[(s * 25 + uu1) * 25 + v1];
    }
  }
}

// ---------------------------------------------------------------------------
// Apply spatial attention: ys[n][s*64+c][t][v] = sum_u x[n][c][t][u]*att[n][s][u][v]
// M = (c,t) = 2048, K = 25 pad 32; 2 m-tiles x 2 v-tiles per wave.
// ---------------------------------------------------------------------------
__global__ __launch_bounds__(32) void apply_att_s(
    const float* __restrict__ x, const float* __restrict__ att,
    float* __restrict__ ys) {
  const int lane = threadIdx.x & 31, half = lane >> 4, lm = lane & 15;
  const int mt = blockIdx.x * 2;
  const int n = blockIdx.y / 3, s = blockIdx.y % 3;
  const float* xN = x + (size_t)n * 64 * 800;
  const float* ab = att + ((size_t)n * 3 + s) * 625;
  const int v0 = lm, v1 = 16 + lm;
  const int v1c = v1 < 25 ? v1 : 24;
  const bool v1ok = v1 < 25;
  v16bf a0, a1, b0, b1;
  {
    int m0 = mt * 16 + lm, m1 = m0 + 16;
    const float* xr0 = xN + (size_t)(m0 >> 5) * 800 + (m0 & 31) * 25;
    const float* xr1 = xN + (size_t)(m1 >> 5) * 800 + (m1 & 31) * 25;
#pragma unroll
    for (int e = 0; e < 16; ++e) {
      int k = a_kmap(e, half);
      int kc = k < 25 ? k : 24;
      float x0v = xr0[kc], x1v = xr1[kc];
      a0[e] = (__bf16)(k < 25 ? x0v : 0.f);
      a1[e] = (__bf16)(k < 25 ? x1v : 0.f);
    }
  }
#pragma unroll
  for (int e = 0; e < 16; ++e) {
    int k = a_kmap(e, half);
    const float* ar = ab + (k < 25 ? k : 24) * 25;
    float av0 = ar[v0], av1 = ar[v1c];
    bool kok = k < 25;
    b0[e] = (__bf16)(kok ? av0 : 0.f);
    b1[e] = (__bf16)((kok && v1ok) ? av1 : 0.f);
  }
  v8f acc00 = {}, acc01 = {}, acc10 = {}, acc11 = {};
  acc00 = wmma_bf16(a0, b0, acc00);
  acc01 = wmma_bf16(a0, b1, acc01);
  acc10 = wmma_bf16(a1, b0, acc10);
  acc11 = wmma_bf16(a1, b1, acc11);
  float* ysN = ys + (size_t)n * 192 * 800 + (size_t)s * 64 * 800;
#pragma unroll
  for (int r = 0; r < 8; ++r) {
    int m0 = mt * 16 + half * 8 + r, m1 = m0 + 16;
    float* yr0 = ysN + (size_t)(m0 >> 5) * 800 + (m0 & 31) * 25;
    float* yr1 = ysN + (size_t)(m1 >> 5) * 800 + (m1 & 31) * 25;
    yr0[v0] = acc00[r];
    yr1[v0] = acc10[r];
    if (v1ok) {
      yr0[v1] = acc01[r];
      yr1[v1] = acc11[r];
    }
  }
}

// ---------------------------------------------------------------------------
// Temporal attention: att_X[n][s][t][q] = (sum_{c,v} qX*kX)/400 * alpha * mask
// K = (c,v) = 400 pad 416; one wave does the whole 32x32 output (4 WMMAs/step).
// X: 0=fwd(alphat_1, mask q<=t), 1=bwd(alphat_0, mask q>=t), 2=ctx(alphat_2).
// ---------------------------------------------------------------------------
__global__ __launch_bounds__(32) void att_temporal(
    const float* __restrict__ qk, const float* __restrict__ a0p,
    const float* __restrict__ a1p, const float* __restrict__ a2p,
    float* __restrict__ att) {
  const int lane = threadIdx.x & 31, half = lane >> 4, lm = lane & 15;
  const int zb = blockIdx.x;
  const int X = zb % 3, s = (zb / 3) % 3, n = zb / 9;
  const float* qb = qk + ((size_t)n * 288 + X * 48 + s * 16) * 800;
  const float* kb = qk + ((size_t)n * 288 + (3 + X) * 48 + s * 16) * 800;
  const int t0 = lm, t1 = 16 + lm, q0 = lm, q1 = 16 + lm;
  v8f acc00 = {}, acc01 = {}, acc10 = {}, acc11 = {};
  for (int k0 = 0; k0 < 416; k0 += 32) {
    v16bf a0, a1, b0, b1;
#pragma unroll
    for (int e = 0; e < 16; ++e) {
      int Ki = k0 + a_kmap(e, half);
      int Kc = Ki < 400 ? Ki : 399;
      int c = Kc / 25, v = Kc - c * 25;
      const float* qr = qb + (size_t)c * 800 + v;
      float qa = qr[t0 * 25], qbv = qr[t1 * 25];
      bool ok = Ki < 400;
      a0[e] = (__bf16)(ok ? qa : 0.f);
      a1[e] = (__bf16)(ok ? qbv : 0.f);
    }
#pragma unroll
    for (int e = 0; e < 16; ++e) {
      int Ki = k0 + a_kmap(e, half);
      int Kc = Ki < 400 ? Ki : 399;
      int c = Kc / 25, v = Kc - c * 25;
      const float* kr = kb + (size_t)c * 800 + v;
      float k0v = kr[q0 * 25], k1v = kr[q1 * 25];
      bool ok = Ki < 400;
      b0[e] = (__bf16)(ok ? k0v : 0.f);
      b1[e] = (__bf16)(ok ? k1v : 0.f);
    }
    acc00 = wmma_bf16(a0, b0, acc00);
    acc01 = wmma_bf16(a0, b1, acc01);
    acc10 = wmma_bf16(a1, b0, acc10);
    acc11 = wmma_bf16(a1, b1, acc11);
  }
  float alpha = (X == 0) ? a1p[s] : (X == 1 ? a0p[s] : a2p[s]);
  float* ab = att + (size_t)zb * 1024;
#pragma unroll
  for (int r = 0; r < 8; ++r) {
    int ta = half * 8 + r, tb = 16 + ta;
    float m00 = 1.f, m01 = 1.f, m10 = 1.f, m11 = 1.f;
    if (X == 0) {  // forward: q <= t
      m00 = (q0 <= ta) ? 1.f : 0.f; m01 = (q1 <= ta) ? 1.f : 0.f;
      m10 = (q0 <= tb) ? 1.f : 0.f; m11 = (q1 <= tb) ? 1.f : 0.f;
    } else if (X == 1) {  // backward: q >= t
      m00 = (q0 >= ta) ? 1.f : 0.f; m01 = (q1 >= ta) ? 1.f : 0.f;
      m10 = (q0 >= tb) ? 1.f : 0.f; m11 = (q1 >= tb) ? 1.f : 0.f;
    }
    ab[ta * 32 + q0] = acc00[r] * (1.f / 400.f) * alpha * m00;
    ab[ta * 32 + q1] = acc01[r] * (1.f / 400.f) * alpha * m01;
    ab[tb * 32 + q0] = acc10[r] * (1.f / 400.f) * alpha * m10;
    ab[tb * 32 + q1] = acc11[r] * (1.f / 400.f) * alpha * m11;
  }
}

// ---------------------------------------------------------------------------
// Apply temporal attention: zcat[n][X*192+s*64+c][q][v] = sum_t att[t][q]*y[n][c][t][v]
// A = att^T (32x32); 2 q-tiles x 2 column-tiles per wave.
// ---------------------------------------------------------------------------
__global__ __launch_bounds__(32) void apply_att_t(
    const float* __restrict__ y, const float* __restrict__ att,
    float* __restrict__ zcat) {
  const int lane = threadIdx.x & 31, half = lane >> 4, lm = lane & 15;
  const int jt = blockIdx.x * 2;
  const int zb = blockIdx.y;
  const int X = zb % 3, s = (zb / 3) % 3, n = zb / 9;
  const float* ab = att + (size_t)zb * 1024;
  const float* yN = y + (size_t)n * 64 * 800;
  const int q0 = lm, q1 = 16 + lm;
  const int j0 = jt * 16 + lm, j1 = j0 + 16;
  const int c0 = j0 / 25, v0 = j0 - c0 * 25;
  const int c1 = j1 / 25, v1 = j1 - c1 * 25;
  v16bf a0, a1, b0, b1;
#pragma unroll
  for (int e = 0; e < 16; ++e) {
    int t = a_kmap(e, half);
    a0[e] = (__bf16)ab[t * 32 + q0];  // A[q][t] = att[t][q]
    a1[e] = (__bf16)ab[t * 32 + q1];
  }
#pragma unroll
  for (int e = 0; e < 16; ++e) {
    int t = a_kmap(e, half);
    b0[e] = (__bf16)yN[(size_t)c0 * 800 + t * 25 + v0];
    b1[e] = (__bf16)yN[(size_t)c1 * 800 + t * 25 + v1];
  }
  v8f acc00 = {}, acc01 = {}, acc10 = {}, acc11 = {};
  acc00 = wmma_bf16(a0, b0, acc00);
  acc01 = wmma_bf16(a0, b1, acc01);
  acc10 = wmma_bf16(a1, b0, acc10);
  acc11 = wmma_bf16(a1, b1, acc11);
  float* zbase = zcat + (size_t)n * 576 * 800 + (size_t)(X * 192 + s * 64) * 800;
  float* zp0 = zbase + (size_t)c0 * 800 + v0;
  float* zp1 = zbase + (size_t)c1 * 800 + v1;
#pragma unroll
  for (int r = 0; r < 8; ++r) {
    int qa = half * 8 + r;
    zp0[qa * 25] = acc00[r];
    zp1[qa * 25] = acc01[r];
    zp0[(16 + qa) * 25] = acc10[r];
    zp1[(16 + qa) * 25] = acc11[r];
  }
}

// ---------------------------------------------------------------------------
// 7x1 temporal conv (pad 3) + bias + BN + residual lrelu -> final output.
// K = (kt,ci) = 448 -> 14 bf16 steps; 2 o-tiles x 2 p-tiles per wave.
// ---------------------------------------------------------------------------
__global__ __launch_bounds__(32) void ext_conv(
    const float* __restrict__ z, const float* __restrict__ w,
    const float* __restrict__ bias,
    const float* __restrict__ bn_g, const float* __restrict__ bn_b,
    const float* __restrict__ bn_m, const float* __restrict__ bn_v,
    float* __restrict__ out) {
  const int lane = threadIdx.x & 31, half = lane >> 4, lm = lane & 15;
  const int pt = blockIdx.x * 2, ob = blockIdx.y * 32, n = blockIdx.z;
  const int p0 = pt * 16 + lm, p1 = p0 + 16;
  const int ta = p0 / 25, va = p0 - ta * 25;
  const int tb = p1 / 25, vb2 = p1 - tb * 25;
  const float* zN = z + (size_t)n * 64 * 800;
  const float* wrow0 = w + (size_t)(ob + lm) * 448;       // w[o][ci][kt]
  const float* wrow1 = w + (size_t)(ob + 16 + lm) * 448;
  v8f acc00 = {}, acc01 = {}, acc10 = {}, acc11 = {};
  for (int st = 0; st < 14; ++st) {
    const int kt = st >> 1, cbase = (st & 1) * 32;
    const int tsa = ta + kt - 3, tsb = tb + kt - 3;
    const int tca = tsa < 0 ? 0 : (tsa > 31 ? 31 : tsa);  // clamped addr
    const int tcb = tsb < 0 ? 0 : (tsb > 31 ? 31 : tsb);
    const bool oka = (tsa >= 0) && (tsa < 32);
    const bool okb = (tsb >= 0) && (tsb < 32);
    v16bf a0, a1, b0, b1;
#pragma unroll
    for (int e = 0; e < 16; ++e) {
      int ci = cbase + a_kmap(e, half);
      a0[e] = (__bf16)wrow0[ci * 7 + kt];
      a1[e] = (__bf16)wrow1[ci * 7 + kt];
    }
#pragma unroll
    for (int e = 0; e < 16; ++e) {
      int ci = cbase + a_kmap(e, half);
      const float* zr = zN + (size_t)ci * 800;
      float z0v = zr[tca * 25 + va], z1v = zr[tcb * 25 + vb2];
      b0[e] = (__bf16)(oka ? z0v : 0.f);
      b1[e] = (__bf16)(okb ? z1v : 0.f);
    }
    acc00 = wmma_bf16(a0, b0, acc00);
    acc01 = wmma_bf16(a0, b1, acc01);
    acc10 = wmma_bf16(a1, b0, acc10);
    acc11 = wmma_bf16(a1, b1, acc11);
  }
  float* outN = out + (size_t)n * 64 * 800;
#pragma unroll
  for (int r = 0; r < 8; ++r) {
    int o0 = ob + half * 8 + r, o1 = o0 + 16;
    float inv0 = bn_g[o0] * rsqrtf(bn_v[o0] + 1e-5f);
    float add0 = bn_b[o0] - bn_m[o0] * inv0;
    float inv1 = bn_g[o1] * rsqrtf(bn_v[o1] + 1e-5f);
    float add1 = bn_b[o1] - bn_m[o1] * inv1;
    float x00 = (acc00[r] + bias[o0]) * inv0 + add0 + zN[(size_t)o0 * 800 + p0];
    float x01 = (acc01[r] + bias[o0]) * inv0 + add0 + zN[(size_t)o0 * 800 + p1];
    float x10 = (acc10[r] + bias[o1]) * inv1 + add1 + zN[(size_t)o1 * 800 + p0];
    float x11 = (acc11[r] + bias[o1]) * inv1 + add1 + zN[(size_t)o1 * 800 + p1];
    outN[(size_t)o0 * 800 + p0] = x00 > 0.f ? x00 : 0.1f * x00;
    outN[(size_t)o0 * 800 + p1] = x01 > 0.f ? x01 : 0.1f * x01;
    outN[(size_t)o1 * 800 + p0] = x10 > 0.f ? x10 : 0.1f * x10;
    outN[(size_t)o1 * 800 + p1] = x11 > 0.f ? x11 : 0.1f * x11;
  }
}

// ---------------------------------------------------------------------------
extern "C" void kernel_launch(void* const* d_in, const int* in_sizes, int n_in,
                              void* d_out, int out_size, void* d_ws, size_t ws_size,
                              hipStream_t stream) {
  const float* x        = (const float*)d_in[0];
  const float* w_in_s   = (const float*)d_in[1];
  const float* b_in_s   = (const float*)d_in[2];
  const float* alphas   = (const float*)d_in[3];
  const float* att0s    = (const float*)d_in[4];
  const float* w_out_s  = (const float*)d_in[5];
  const float* b_out_s  = (const float*)d_in[6];
  const float* g_out_s  = (const float*)d_in[7];
  const float* bt_out_s = (const float*)d_in[8];
  const float* m_out_s  = (const float*)d_in[9];
  const float* v_out_s  = (const float*)d_in[10];
  const float* w_ff_s   = (const float*)d_in[11];
  const float* b_ff_s   = (const float*)d_in[12];
  const float* g_ff_s   = (const float*)d_in[13];
  const float* bt_ff_s  = (const float*)d_in[14];
  const float* m_ff_s   = (const float*)d_in[15];
  const float* v_ff_s   = (const float*)d_in[16];
  const float* w_in_t   = (const float*)d_in[17];
  const float* b_in_t   = (const float*)d_in[18];
  const float* at0      = (const float*)d_in[19];
  const float* at1      = (const float*)d_in[20];
  const float* at2      = (const float*)d_in[21];
  const float* w_out_t  = (const float*)d_in[22];
  const float* b_out_t  = (const float*)d_in[23];
  const float* g_out_t  = (const float*)d_in[24];
  const float* bt_out_t = (const float*)d_in[25];
  const float* m_out_t  = (const float*)d_in[26];
  const float* v_out_t  = (const float*)d_in[27];
  const float* w_ff_t   = (const float*)d_in[28];
  const float* b_ff_t   = (const float*)d_in[29];
  const float* g_ff_t   = (const float*)d_in[30];
  const float* bt_ff_t  = (const float*)d_in[31];
  const float* m_ff_t   = (const float*)d_in[32];
  const float* v_ff_t   = (const float*)d_in[33];
  const float* w_ext    = (const float*)d_in[34];
  const float* b_ext    = (const float*)d_in[35];
  const float* g_ext    = (const float*)d_in[36];
  const float* bt_ext   = (const float*)d_in[37];
  const float* m_ext    = (const float*)d_in[38];
  const float* v_ext    = (const float*)d_in[39];
  (void)in_sizes; (void)n_in; (void)out_size; (void)ws_size;

  // Workspace arena (float offsets), lifetime-based reuse:
  //   y1   [0, 13107200)           alive through temporal block
  //   y0   [13107200, 26214400)
  //   ysb  [26214400, 65536000)
  //   qks  [65536000, 85196800)
  //   atts [85196800, 85676800)
  //   qkt  [13107200, 72089600)    reuse (y0/ysb/qks dead)
  //   zcat [13107200, 131072000)   reuse (qkt dead)
  //   attt [131072000, 133431296)
  //   z0   [133431296, 146538496)
  //   z1   [13107200, 26214400)    reuse (zcat dead)
  //   pe_s [146538496, 146540096)  persistent
  //   pe_t [146540096, 146542144)  persistent
  float* ws   = (float*)d_ws;
  float* y1   = ws;
  float* y0   = ws + 13107200ull;
  float* ysb  = ws + 26214400ull;
  float* qks  = ws + 65536000ull;
  float* atts = ws + 85196800ull;
  float* qkt  = ws + 13107200ull;
  float* zcat = ws + 13107200ull;
  float* attt = ws + 131072000ull;
  float* z0   = ws + 133431296ull;
  float* z1   = ws + 13107200ull;
  float* pe_s = ws + 146538496ull;
  float* pe_t = ws + 146540096ull;
  float* outp = (float*)d_out;

  dim3 blk(32);
  make_pe<<<dim3(8), dim3(256), 0, stream>>>(pe_s, pe_t);
  // Spatial block
  conv1x1_wmma<1, 0><<<dim3(25, 3, 256), blk, 0, stream>>>(
      x, w_in_s, b_in_s, nullptr, nullptr, nullptr, nullptr, nullptr, pe_s,
      qks, 64, 96);
  att_spatial<<<dim3(768), blk, 0, stream>>>(qks, alphas, att0s, atts);
  apply_att_s<<<dim3(64, 768), blk, 0, stream>>>(x, atts, ysb);
  conv1x1_wmma<0, 1><<<dim3(25, 2, 256), blk, 0, stream>>>(
      ysb, w_out_s, b_out_s, x, g_out_s, bt_out_s, m_out_s, v_out_s, nullptr,
      y0, 192, 64);
  conv1x1_wmma<0, 1><<<dim3(25, 2, 256), blk, 0, stream>>>(
      y0, w_ff_s, b_ff_s, x, g_ff_s, bt_ff_s, m_ff_s, v_ff_s, nullptr,
      y1, 64, 64);
  // Temporal block
  conv1x1_wmma<2, 0><<<dim3(25, 9, 256), blk, 0, stream>>>(
      y1, w_in_t, b_in_t, nullptr, nullptr, nullptr, nullptr, nullptr, pe_t,
      qkt, 64, 288);
  att_temporal<<<dim3(2304), blk, 0, stream>>>(qkt, at0, at1, at2, attt);
  apply_att_t<<<dim3(50, 2304), blk, 0, stream>>>(y1, attt, zcat);
  conv1x1_wmma<0, 1><<<dim3(25, 2, 256), blk, 0, stream>>>(
      zcat, w_out_t, b_out_t, y1, g_out_t, bt_out_t, m_out_t, v_out_t, nullptr,
      z0, 576, 64);
  conv1x1_wmma<0, 1><<<dim3(25, 2, 256), blk, 0, stream>>>(
      z0, w_ff_t, b_ff_t, y1, g_ff_t, bt_ff_t, m_ff_t, v_ff_t, nullptr,
      z1, 64, 64);
  // 7x1 temporal conv + BN + residual lrelu -> output  (50 p-tiles, 2/wave)
  ext_conv<<<dim3(25, 2, 256), blk, 0, stream>>>(
      z1, w_ext, b_ext, g_ext, bt_ext, m_ext, v_ext, outp);
}
